// MinGRU_13065290514805
// MI455X (gfx1250) — matrix operand order repaired and checked by reading, not compiled, not run on hardware
//
#include <hip/hip_runtime.h>

typedef __attribute__((ext_vector_type(16))) _Float16 v16h;
typedef __attribute__((ext_vector_type(8)))  _Float16 v8h;
typedef __attribute__((ext_vector_type(4)))  _Float16 v4h;
typedef __attribute__((ext_vector_type(8)))  float    v8f;

constexpr int BATCH = 8;
constexpr int SEQ   = 4096;
constexpr int IN    = 512;   // K dimension
constexpr int HID   = 512;   // N dimension
constexpr int ROW   = IN + 8; // padded LDS row (halves) to break bank conflicts

// One wave owns 16 hidden columns for one batch element and walks all of S.
// Per 16-timestep chunk: 2 projections x 16 WMMA(16x16x32 f16) with f32 acc,
// then a 16-step serial scan done by 16 lanes via a small LDS transpose buffer.
__global__ __launch_bounds__(64)
void mingru_fused_wmma(const float* __restrict__ x,
                       const float* __restrict__ h0,
                       const float* __restrict__ Wz,
                       const float* __restrict__ bz,
                       const float* __restrict__ Wh,
                       const float* __restrict__ bh,
                       float* __restrict__ out)
{
    __shared__ _Float16 lwz[2][16][ROW];   // per-wave W_z tile, f16, row = h col
    __shared__ _Float16 lwh[2][16][ROW];   // per-wave W_h tile
    __shared__ float    sa[2][16][17];     // (1 - z)   [time][hcol]
    __shared__ float    sbv[2][16][17];    // z * h~    [time][hcol]

    const int lane = threadIdx.x & 31;
    const int wave = threadIdx.x >> 5;
    const int b     = blockIdx.x >> 4;                       // batch
    const int hbase = ((blockIdx.x & 15) << 5) + (wave << 4); // 16-col tile
    const int n = lane & 15;   // column within tile (A-row m / B-col n / D-col N)
    const int g = lane >> 4;   // lane group (selects K sub-range / D-row half)

    // ---- Stage this wave's weight rows into LDS as f16 (once) ----
    for (int r = 0; r < 16; ++r) {
        const float* wzrow = Wz + (size_t)(hbase + r) * IN;
        const float* whrow = Wh + (size_t)(hbase + r) * IN;
        for (int k = lane * 4; k < IN; k += 128) {
            float4 fz = *(const float4*)(wzrow + k);
            float4 fh = *(const float4*)(whrow + k);
            v4h hz = {(_Float16)fz.x, (_Float16)fz.y, (_Float16)fz.z, (_Float16)fz.w};
            v4h hh = {(_Float16)fh.x, (_Float16)fh.y, (_Float16)fh.z, (_Float16)fh.w};
            *(v4h*)&lwz[wave][r][k] = hz;
            *(v4h*)&lwh[wave][r][k] = hh;
        }
    }
    __syncthreads();

    const float bzv = bz[hbase + n];
    const float bhv = bh[hbase + n];
    float hcar = h0[(size_t)b * HID + hbase + n];   // valid in lanes 0..15

    // lane's A-matrix row m == n (time s0+m); advance 16 rows per chunk
    const float* xrow = x + ((size_t)b * SEQ + n) * IN;

    for (int s0 = 0; s0 < SEQ; s0 += 16) {
        if (s0 + 16 < SEQ)
            __builtin_prefetch(xrow + 16 * IN, 0, 0);   // global_prefetch_b8

        v8f cz = {};
        v8f ch = {};
        #pragma unroll 4
        for (int c = 0; c < 16; ++c) {
            // A fragment: lane holds k = c*32 + g*8 + [0..7] and +16..+23
            const float* p = xrow + c * 32 + g * 8;
            float4 f0 = *(const float4*)(p);
            float4 f1 = *(const float4*)(p + 4);
            float4 f2 = *(const float4*)(p + 16);
            float4 f3 = *(const float4*)(p + 20);
            v16h a;
            a[0]=(_Float16)f0.x;  a[1]=(_Float16)f0.y;  a[2]=(_Float16)f0.z;  a[3]=(_Float16)f0.w;
            a[4]=(_Float16)f1.x;  a[5]=(_Float16)f1.y;  a[6]=(_Float16)f1.z;  a[7]=(_Float16)f1.w;
            a[8]=(_Float16)f2.x;  a[9]=(_Float16)f2.y;  a[10]=(_Float16)f2.z; a[11]=(_Float16)f2.w;
            a[12]=(_Float16)f3.x; a[13]=(_Float16)f3.y; a[14]=(_Float16)f3.z; a[15]=(_Float16)f3.w;

            // B fragments from LDS: lane = col n, k = c*32 + g*8 + [0..7] and +16..+23
            const int ko = c * 32 + g * 8;
            v8h z0 = *(const v8h*)&lwz[wave][n][ko];
            v8h z1 = *(const v8h*)&lwz[wave][n][ko + 16];
            v8h w0 = *(const v8h*)&lwh[wave][n][ko];
            v8h w1 = *(const v8h*)&lwh[wave][n][ko + 16];
            v16h bmz, bmh;
            #pragma unroll
            for (int i = 0; i < 8; ++i) {
                bmz[i] = z0[i]; bmz[i + 8] = z1[i];
                bmh[i] = w0[i]; bmh[i + 8] = w1[i];
            }

            cz = __builtin_amdgcn_wmma_f32_16x16x32_f16(false, a, false, bmz,
                                                        (short)0, cz, false, false);
            ch = __builtin_amdgcn_wmma_f32_16x16x32_f16(false, a, false, bmh,
                                                        (short)0, ch, false, false);
        }

        // Epilogue: gate + candidate, drop (a,b) into LDS in [time][col] order.
        // D layout: VGPR r, lanes 0-15 -> M=r, lanes 16-31 -> M=r+8; N = lane%16.
        #pragma unroll
        for (int r = 0; r < 8; ++r) {
            const int t = r + (g << 3);
            const float zg   = 1.0f / (1.0f + __expf(-(cz[r] + bzv)));
            const float cand = ch[r] + bhv;
            sa [wave][t][n] = 1.0f - zg;
            sbv[wave][t][n] = zg * cand;
        }
        __syncthreads();

        // Serial scan over 16 timesteps: lane j owns hidden column hbase+j.
        if (lane < 16) {
            float hh = hcar;
            float* orow = out + ((size_t)b * SEQ + s0) * HID + hbase + lane;
            #pragma unroll
            for (int t = 0; t < 16; ++t) {
                hh = sa[wave][t][lane] * hh + sbv[wave][t][lane];
                orow[(size_t)t * HID] = hh;
            }
            hcar = hh;
        }
        xrow += 16 * IN;
    }
}

extern "C" void kernel_launch(void* const* d_in, const int* in_sizes, int n_in,
                              void* d_out, int out_size, void* d_ws, size_t ws_size,
                              hipStream_t stream) {
    (void)in_sizes; (void)n_in; (void)d_ws; (void)ws_size; (void)out_size;
    const float* x  = (const float*)d_in[0];
    const float* h0 = (const float*)d_in[1];
    const float* Wz = (const float*)d_in[2];
    const float* bz = (const float*)d_in[3];
    const float* Wh = (const float*)d_in[4];
    const float* bh = (const float*)d_in[5];
    float* out = (float*)d_out;

    dim3 grid(BATCH * (HID / 32));   // 128 blocks: (batch, 32-col h-block)
    dim3 block(64);                  // 2 waves, each owning 16 hidden columns
    mingru_fused_wmma<<<grid, block, 0, stream>>>(x, h0, Wz, bz, Wh, bh, out);
}
